// VariationalAutoEncoder_58291296141330
// MI455X (gfx1250) — matrix operand order repaired and checked by reading, not compiled, not run on hardware
//
#include <hip/hip_runtime.h>
#include <hip/hip_bf16.h>
#include <stdint.h>

// ---------------- problem constants ----------------
#define Nn      65536      // nodes
#define Bn      128        // graphs
#define NPG     512
#define En      524288     // edges
#define D_IN    128
#define Hh      256
#define LAT     64
#define HD      256
#define NMAX    512
#define Pp      130816     // NMAX*(NMAX-1)/2
#define BN_EPS  1e-5f
#define LEAKY   0.2f

typedef __bf16 bf16_t;
typedef __attribute__((ext_vector_type(16))) __bf16 v16bf;
typedef __attribute__((ext_vector_type(8)))  float  v8f;

union FragU { v16bf v; uint32_t u[8]; };

// native device-scope fp32 atomic add (lowers to global_atomic_add_f32, no CAS loop)
__device__ __forceinline__ void atomAddF32(float* p, float v) {
    __hip_atomic_fetch_add(p, v, __ATOMIC_RELAXED, __HIP_MEMORY_SCOPE_AGENT);
}

// ------------- fp32 -> bf16 convert (with optional K padding) -------------
__global__ void k_cvt_pad(const float* __restrict__ in, bf16_t* __restrict__ out,
                          long long rows, int Kin, int Kout) {
    long long idx = (long long)blockIdx.x * blockDim.x + threadIdx.x;
    long long total = rows * (long long)Kout;
    if (idx >= total) return;
    int k = (int)(idx % Kout);
    long long r = idx / Kout;
    out[idx] = (k < Kin) ? (bf16_t)in[r * Kin + k] : (bf16_t)0.0f;
}

// ------------- BN-affine convert: out = bf16(a[col]*Z + b[col]) -------------
__global__ void k_cvt_affine(const float* __restrict__ Z, const float* __restrict__ a,
                             const float* __restrict__ b, bf16_t* __restrict__ out,
                             long long total, int C) {
    long long idx = (long long)blockIdx.x * blockDim.x + threadIdx.x;
    if (idx >= total) return;
    int c = (int)(idx % C);
    out[idx] = (bf16_t)(a[c] * Z[idx] + b[c]);
}

// ------------- edge scatter-add: agg[dst] += h[src] (L2-resident fp32 atomics) -------------
__global__ void k_seg_add(const float* __restrict__ h, float* __restrict__ agg,
                          const int* __restrict__ src, const int* __restrict__ dst,
                          int E, int D) {
    int w = (blockIdx.x * blockDim.x + threadIdx.x) >> 5;
    if (w >= E) return;
    int lane = threadIdx.x & 31;
    const float* hp = h   + (size_t)src[w] * D;
    float*       ap = agg + (size_t)dst[w] * D;
    for (int c = lane; c < D; c += 32) atomAddF32(&ap[c], hp[c]);
}

// ------------- big WMMA GEMM: C[M,Cout] = leaky(A[M,K] @ W[Cout,K]^T + bias) -------------
// optional fused per-column sum/sumsq (for BN) via atomics
__global__ void k_gemm_bf16(const bf16_t* __restrict__ A, int K,
                            const bf16_t* __restrict__ W,
                            const float* __restrict__ bias,
                            float* __restrict__ C, int Cout, int M,
                            float* colsum, float* colsumsq) {
    int wid  = (blockIdx.x * blockDim.x + threadIdx.x) >> 5;
    int lane = threadIdx.x & 31;
    int tilesN = Cout >> 4;
    int tn = wid % tilesN, tm = wid / tilesN;
    if (tm * 16 >= M) return;
    int n = lane & 15, grp = lane >> 4;

    const uint32_t* arow = (const uint32_t*)(A + (size_t)(tm * 16 + n) * K);
    const uint32_t* brow = (const uint32_t*)(W + (size_t)(tn * 16 + n) * K);

    v8f acc = (v8f)(0.0f);
    for (int k0 = 0; k0 < K; k0 += 32) {
        int kd = k0 >> 1;              // dword index within row
        FragU a, b;
#pragma unroll
        for (int v = 0; v < 4; ++v) {  // A 16x32: lanes0-15 K=[0,8)+[16,24), lanes16-31 +8
            a.u[v]     = arow[kd + grp * 4 + v];
            a.u[4 + v] = arow[kd + 8 + grp * 4 + v];
        }
#pragma unroll
        for (int v = 0; v < 8; ++v)    // B 32x16: lanes0-15 K=[0,8)+[16,24), lanes16-31 +8
            b.u[v] = brow[kd + grp * 4 + (v & 3) + (v >> 2) * 8];
        acc = __builtin_amdgcn_wmma_f32_16x16x32_bf16(false, a.v, false, b.v,
                                                      (short)0, acc, false, false);
    }
    int col = tn * 16 + n;
    float bia = bias ? bias[col] : 0.0f;
    float s = 0.f, s2 = 0.f;
    float* cp = C + (size_t)(tm * 16 + grp * 8) * Cout + col;
#pragma unroll
    for (int r = 0; r < 8; ++r) {
        float v = acc[r] + bia;
        v = v > 0.f ? v : LEAKY * v;
        cp[(size_t)r * Cout] = v;
        s += v; s2 += v * v;
    }
    if (colsum) {
        s  += __shfl_xor(s, 16, 32);
        s2 += __shfl_xor(s2, 16, 32);
        if (grp == 0) { atomAddF32(&colsum[col], s); atomAddF32(&colsumsq[col], s2); }
    }
}

// ------------- BN finalize: a = g*rsqrt(var+eps), b = be - a*mean -------------
__global__ void k_bn_finalize(const float* colsum, const float* colsumsq,
                              const float* g, const float* be,
                              float* aOut, float* bOut, float invN, int C) {
    int c = blockIdx.x * blockDim.x + threadIdx.x;
    if (c >= C) return;
    float m = colsum[c] * invN;
    float v = colsumsq[c] * invN - m * m;
    float a = g[c] * rsqrtf(v + BN_EPS);
    aOut[c] = a;
    bOut[c] = be[c] - a * m;
}

// ------------- per-graph pooled segment sum + concat stats -------------
__global__ void k_pool(const float* __restrict__ h, const float* __restrict__ stats,
                       float* __restrict__ outcat) {
    int g = blockIdx.x, c = threadIdx.x;           // 256 threads
    float s = 0.f;
    const float* hp = h + (size_t)g * NPG * Hh + c;
    for (int r = 0; r < NPG; ++r) s += hp[(size_t)r * Hh];
    outcat[g * 263 + c] = s;
    if (c < 7) outcat[g * 263 + 256 + c] = stats[g * 7 + c];
}

// ------------- small BN over B=128 rows, emit padded bf16 [128,288] -------------
__global__ void k_bn_small(const float* __restrict__ outcat, const float* __restrict__ g,
                           const float* __restrict__ b, bf16_t* __restrict__ Afc) {
    int c = threadIdx.x;                            // 288 threads
    if (c < 263) {
        float m = 0.f, v = 0.f;
        for (int r = 0; r < Bn; ++r) { float x = outcat[r * 263 + c]; m += x; v += x * x; }
        m *= (1.0f / Bn); v = v * (1.0f / Bn) - m * m;
        float a  = g[c] * rsqrtf(v + BN_EPS);
        float sh = b[c] - a * m;
        for (int r = 0; r < Bn; ++r)
            Afc[r * 288 + c] = (bf16_t)(a * outcat[r * 263 + c] + sh);
    } else {
        for (int r = 0; r < Bn; ++r) Afc[r * 288 + c] = (bf16_t)0.0f;
    }
}

// ------------- small WMMA GEMM, M=128: one block per N-tile, 8 waves = 8 M-tiles -------------
__global__ void k_small_gemm(const bf16_t* __restrict__ A, int K,
                             const bf16_t* __restrict__ W, const float* __restrict__ bias,
                             int Cout, int act, float* outF, bf16_t* outB) {
    int tm = threadIdx.x >> 5;       // 0..7  (M tiles)
    int tn = blockIdx.x;             // N tile
    int lane = threadIdx.x & 31;
    int n = lane & 15, grp = lane >> 4;
    const uint32_t* arow = (const uint32_t*)(A + (size_t)(tm * 16 + n) * K);
    const uint32_t* brow = (const uint32_t*)(W + (size_t)(tn * 16 + n) * K);
    v8f acc = (v8f)(0.0f);
    for (int k0 = 0; k0 < K; k0 += 32) {
        int kd = k0 >> 1;
        FragU a, b;
#pragma unroll
        for (int v = 0; v < 4; ++v) {
            a.u[v]     = arow[kd + grp * 4 + v];
            a.u[4 + v] = arow[kd + 8 + grp * 4 + v];
        }
#pragma unroll
        for (int v = 0; v < 8; ++v) b.u[v] = brow[kd + grp * 4 + (v & 3) + (v >> 2) * 8];
        acc = __builtin_amdgcn_wmma_f32_16x16x32_bf16(false, a.v, false, b.v,
                                                      (short)0, acc, false, false);
    }
    int col = tn * 16 + n;
    float bia = bias ? bias[col] : 0.0f;
#pragma unroll
    for (int r = 0; r < 8; ++r) {
        float v = acc[r] + bia;
        if (act == 1) v = fmaxf(v, 0.0f);
        size_t row = (size_t)(tm * 16 + grp * 8 + r);
        if (outF) outF[row * Cout + col] = v;
        if (outB) outB[row * Cout + col] = (bf16_t)v;
    }
}

// ------------- z = mu + exp(0.5*lv)*eps (tiny, fp32) -------------
__global__ void k_z(const float* __restrict__ outfc,
                    const float* __restrict__ muW, const float* __restrict__ mub,
                    const float* __restrict__ lvW, const float* __restrict__ lvb,
                    const float* __restrict__ eps, float* __restrict__ z) {
    int idx = blockIdx.x * blockDim.x + threadIdx.x;   // 128*64
    if (idx >= Bn * LAT) return;
    int b = idx / LAT, j = idx % LAT;
    float mu = mub[j], lv = lvb[j];
    const float* o = outfc + (size_t)b * Hh;
    const float* mw = muW + (size_t)j * Hh;
    const float* lw = lvW + (size_t)j * Hh;
    for (int k = 0; k < Hh; ++k) { mu += o[k] * mw[k]; lv += o[k] * lw[k]; }
    z[idx] = mu + __expf(0.5f * lv) * eps[idx];
}

// ------------- zc = concat(z, stats) -> bf16 padded [128,96] -------------
__global__ void k_build_zc(const float* __restrict__ z, const float* __restrict__ stats,
                           bf16_t* __restrict__ zcb) {
    int idx = blockIdx.x * blockDim.x + threadIdx.x;   // 128*96
    if (idx >= Bn * 96) return;
    int b = idx / 96, c = idx % 96;
    float v = 0.f;
    if (c < LAT) v = z[b * LAT + c];
    else if (c < LAT + 7) v = stats[b * 7 + (c - LAT)];
    zcb[idx] = (bf16_t)v;
}

// ------------- triu index tables -------------
__global__ void k_fill_triu(int* iu0, int* iu1) {
    int i = threadIdx.x;                                // 512 threads
    if (i >= NMAX) return;
    int p = i * (NMAX - 1) - (i * (i - 1)) / 2;
    for (int j = i + 1; j < NMAX; ++j) { iu0[p] = i; iu1[p] = j; ++p; }
}

// ------------- dec layer 3: [128,2P] GEMM fused with gumbel-argmax scatter ---------
// Each wave owns one 16-col tile and ALL 8 M-tiles (weights streamed once).
__global__ void k_dec2(const bf16_t* __restrict__ A /*128x256*/,
                       const bf16_t* __restrict__ W /*2P x 256*/,
                       const float* __restrict__ bias,
                       const float* __restrict__ gumbel,
                       const int* __restrict__ iu0, const int* __restrict__ iu1,
                       float* __restrict__ adj) {
    int wid  = (blockIdx.x * blockDim.x + threadIdx.x) >> 5;  // N-tile, 0..16351
    int lane = threadIdx.x & 31;
    int n = lane & 15, grp = lane >> 4;
    const int K = HD;                                          // 256
    v8f acc[8];
#pragma unroll
    for (int i = 0; i < 8; ++i) acc[i] = (v8f)(0.0f);
    const uint32_t* brow  = (const uint32_t*)(W + (size_t)(wid * 16 + n) * K);
    const uint32_t* abase = (const uint32_t*)A;                // row stride = 128 dwords
    for (int ks = 0; ks < 8; ++ks) {
        int kd = ks * 16;
        // give GL2 a head start on the next 64B chunk of the weight stream
        if (ks < 7) __builtin_prefetch(brow + kd + 16 + grp * 8, 0, 1);
        FragU b;
#pragma unroll
        for (int v = 0; v < 8; ++v) b.u[v] = brow[kd + grp * 4 + (v & 3) + (v >> 2) * 8];
#pragma unroll
        for (int mt = 0; mt < 8; ++mt) {
            FragU a;
            const uint32_t* arow = abase + (size_t)(mt * 16 + n) * 128;
#pragma unroll
            for (int v = 0; v < 4; ++v) {
                a.u[v]     = arow[kd + grp * 4 + v];
                a.u[4 + v] = arow[kd + 8 + grp * 4 + v];
            }
            acc[mt] = __builtin_amdgcn_wmma_f32_16x16x32_bf16(false, a.v, false, b.v,
                                                              (short)0, acc[mt], false, false);
        }
    }
    int col  = wid * 16 + n;            // global column in [0, 2P)
    int p    = col >> 1;
    int half = col & 1;
    float bia = bias[col];
    int i0 = iu0[p], j0 = iu1[p];
#pragma unroll
    for (int mt = 0; mt < 8; ++mt) {
#pragma unroll
        for (int r = 0; r < 8; ++r) {
            int b = mt * 16 + grp * 8 + r;            // graph index
            float t = acc[mt][r] + bia + gumbel[((size_t)b * Pp + p) * 2 + half];
            float tp = __shfl_xor(t, 1, 32);          // partner logit (other class)
            if (half == 0) {
                float e = (t >= tp) ? 1.0f : 0.0f;    // argmax (index0 wins ties)
                size_t base = (size_t)b * (NMAX * NMAX);
                adj[base + (size_t)i0 * NMAX + j0] = e;
                adj[base + (size_t)j0 * NMAX + i0] = e;
            }
        }
    }
}

// =====================================================================
extern "C" void kernel_launch(void* const* d_in, const int* in_sizes, int n_in,
                              void* d_out, int out_size, void* d_ws, size_t ws_size,
                              hipStream_t stream) {
    // ---- inputs (setup_inputs dict order, params flattened depth-first) ----
    const float* x      = (const float*)d_in[0];
    const float* stats  = (const float*)d_in[1];
    const float* eps    = (const float*)d_in[2];
    const float* gumbel = (const float*)d_in[3];
    const float *encW1[3], *encb1[3], *encg[3], *encbe[3], *encW2[3], *encb2[3];
    for (int i = 0; i < 3; ++i) {
        int base = 4 + i * 6;
        encW1[i] = (const float*)d_in[base + 0];
        encb1[i] = (const float*)d_in[base + 1];
        encg[i]  = (const float*)d_in[base + 2];
        encbe[i] = (const float*)d_in[base + 3];
        encW2[i] = (const float*)d_in[base + 4];
        encb2[i] = (const float*)d_in[base + 5];
    }
    const float* bn_g  = (const float*)d_in[22];
    const float* bn_b  = (const float*)d_in[23];
    const float* fc_W  = (const float*)d_in[24];
    const float* fc_b  = (const float*)d_in[25];
    const float* mu_W  = (const float*)d_in[26];
    const float* mu_b  = (const float*)d_in[27];
    const float* lv_W  = (const float*)d_in[28];
    const float* lv_b  = (const float*)d_in[29];
    const float* d0W   = (const float*)d_in[30];
    const float* d0b   = (const float*)d_in[31];
    const float* d1W   = (const float*)d_in[32];
    const float* d1b   = (const float*)d_in[33];
    const float* d2W   = (const float*)d_in[34];
    const float* d2b   = (const float*)d_in[35];
    const int*   esrc  = (const int*)d_in[36];
    const int*   edst  = esrc + En;
    float*       adj   = (float*)d_out;

    // ---- workspace bump allocator (256B aligned) ----
    char* ws = (char*)d_ws;
    size_t off = 0;
    auto alloc = [&](size_t bytes) -> char* {
        char* p = ws + off;
        off = (off + bytes + 255) & ~(size_t)255;
        return p;
    };
    float*  agg   = (float*) alloc((size_t)Nn * Hh * 4);
    float*  hbuf  = (float*) alloc((size_t)Nn * Hh * 4);
    bf16_t* Abf   = (bf16_t*)alloc((size_t)Nn * Hh * 2);
    float*  Zbuf  = (float*) alloc((size_t)Nn * Hh * 4);
    bf16_t* Zbf   = (bf16_t*)alloc((size_t)Nn * Hh * 2);
    float*  colsum = (float*)alloc(2 * Hh * 4);        // sum + sumsq (contiguous)
    float*  colsq  = colsum + Hh;
    float*  bnA    = (float*)alloc(Hh * 4);
    float*  bnB    = (float*)alloc(Hh * 4);
    bf16_t* W1b[3], *W2b[3];
    for (int i = 0; i < 3; ++i) { W1b[i] = (bf16_t*)alloc(Hh * Hh * 2);
                                  W2b[i] = (bf16_t*)alloc(Hh * Hh * 2); }
    bf16_t* fcWb  = (bf16_t*)alloc(256 * 288 * 2);
    bf16_t* d0Wb  = (bf16_t*)alloc(256 * 96 * 2);
    bf16_t* d1Wb  = (bf16_t*)alloc(256 * 256 * 2);
    bf16_t* d2Wb  = (bf16_t*)alloc((size_t)2 * Pp * HD * 2);
    float*  outcat= (float*) alloc(Bn * 263 * 4);
    bf16_t* Afc   = (bf16_t*)alloc(Bn * 288 * 2);
    float*  outfc = (float*) alloc(Bn * 256 * 4);
    float*  zbuf  = (float*) alloc(Bn * LAT * 4);
    bf16_t* zcb   = (bf16_t*)alloc(Bn * 96 * 2);
    bf16_t* db0   = (bf16_t*)alloc(Bn * 256 * 2);
    bf16_t* db1   = (bf16_t*)alloc(Bn * 256 * 2);
    int*    iu0   = (int*)   alloc(Pp * 4);
    int*    iu1   = (int*)   alloc(Pp * 4);
    (void)ws_size; (void)n_in; (void)in_sizes; (void)out_size;

    auto cvt = [&](const float* in, bf16_t* out, long long rows, int Kin, int Kout) {
        long long total = rows * (long long)Kout;
        k_cvt_pad<<<(unsigned)((total + 255) / 256), 256, 0, stream>>>(in, out, rows, Kin, Kout);
    };

    // ---- one-time-per-launch weight conversions / tables ----
    for (int i = 0; i < 3; ++i) {
        cvt(encW1[i], W1b[i], Hh, (i == 0 ? D_IN : Hh), (i == 0 ? D_IN : Hh));
        cvt(encW2[i], W2b[i], Hh, Hh, Hh);
    }
    cvt(fc_W, fcWb, 256, 263, 288);
    cvt(d0W, d0Wb, 256, LAT + 7, 96);
    cvt(d1W, d1Wb, 256, 256, 256);
    cvt(d2W, d2Wb, (long long)2 * Pp, HD, HD);
    k_fill_triu<<<1, NMAX, 0, stream>>>(iu0, iu1);
    hipMemsetAsync(adj, 0, (size_t)Bn * NMAX * NMAX * 4, stream);

    // ---- encoder: 3 message-passing + MLP layers ----
    const float* hin = x;
    for (int L = 0; L < 3; ++L) {
        int D = (L == 0) ? D_IN : Hh;
        hipMemcpyAsync(agg, hin, (size_t)Nn * D * 4, hipMemcpyDeviceToDevice, stream);
        k_seg_add<<<En / 8, 256, 0, stream>>>(hin, agg, esrc, edst, En, D);
        cvt(agg, Abf, Nn, D, D);
        hipMemsetAsync(colsum, 0, 2 * Hh * 4, stream);
        // GEMM1: Z = leaky(agg @ W1^T + b1), fused column stats
        k_gemm_bf16<<<(Nn / 16) * (Hh / 16) / 8, 256, 0, stream>>>(
            Abf, D, W1b[L], encb1[L], Zbuf, Hh, Nn, colsum, colsq);
        k_bn_finalize<<<1, Hh, 0, stream>>>(colsum, colsq, encg[L], encbe[L],
                                            bnA, bnB, 1.0f / Nn, Hh);
        k_cvt_affine<<<(unsigned)(((long long)Nn * Hh + 255) / 256), 256, 0, stream>>>(
            Zbuf, bnA, bnB, Zbf, (long long)Nn * Hh, Hh);
        // GEMM2: h = leaky(BN(Z) @ W2^T + b2)
        k_gemm_bf16<<<(Nn / 16) * (Hh / 16) / 8, 256, 0, stream>>>(
            Zbf, Hh, W2b[L], encb2[L], hbuf, Hh, Nn, nullptr, nullptr);
        hin = hbuf;
    }

    // ---- head ----
    k_pool<<<Bn, 256, 0, stream>>>(hbuf, stats, outcat);
    k_bn_small<<<1, 288, 0, stream>>>(outcat, bn_g, bn_b, Afc);
    k_small_gemm<<<256 / 16, 256, 0, stream>>>(Afc, 288, fcWb, fc_b, 256, 0, outfc, nullptr);
    k_z<<<(Bn * LAT + 255) / 256, 256, 0, stream>>>(outfc, mu_W, mu_b, lv_W, lv_b, eps, zbuf);
    k_build_zc<<<(Bn * 96 + 255) / 256, 256, 0, stream>>>(zbuf, stats, zcb);
    k_small_gemm<<<256 / 16, 256, 0, stream>>>(zcb, 96, d0Wb, d0b, 256, 1, nullptr, db0);
    k_small_gemm<<<256 / 16, 256, 0, stream>>>(db0, 256, d1Wb, d1b, 256, 1, nullptr, db1);
    // ---- fused dec2 GEMM + gumbel argmax + adjacency scatter ----
    k_dec2<<<(2 * Pp / 16) / 8, 256, 0, stream>>>(db1, d2Wb, d2b, gumbel, iu0, iu1, adj);
}
